// MultiHeadSelfAttention_63660005261811
// MI455X (gfx1250) — matrix operand order repaired and checked by reading, not compiled, not run on hardware
//
#include <hip/hip_runtime.h>
#include <hip/hip_bf16.h>

// ---------------------------------------------------------------------------
// MHA forward  B=2, T=2048, D=1024, H=16, DK=64  (fp32 in/out, bf16 WMMA)
// x + weights pre-converted to bf16 (weights transposed to [n][k]); GEMM, Q and K
// staging is pure 16B async copies (ASYNCcnt); softmax scale folded into Q.
// ---------------------------------------------------------------------------

typedef __attribute__((ext_vector_type(16))) __bf16 v16bf;
typedef __attribute__((ext_vector_type(8)))  __bf16 v8bf;
typedef __attribute__((ext_vector_type(8)))  float  v8f;

__device__ __forceinline__ __bf16 to_bf16(float f) {
    unsigned u = __builtin_bit_cast(unsigned, f);
    unsigned r = u + 0x7FFFu + ((u >> 16) & 1u);   // round-to-nearest-even
    unsigned short h = (unsigned short)(r >> 16);
    return __builtin_bit_cast(__bf16, h);
}

__device__ __forceinline__ v16bf ld_frag(const __bf16* p0, const __bf16* p1) {
    union { v16bf v; v8bf h[2]; } u;
    u.h[0] = *(const v8bf*)p0;
    u.h[1] = *(const v8bf*)p1;
    return u.v;
}

__device__ __forceinline__ v8f wmma_bf16(v16bf a, v16bf b, v8f c) {
    return __builtin_amdgcn_wmma_f32_16x16x32_bf16(
        false, a, false, b, (short)0, c, false, false);
}

// CDNA5 async copy: 16B global -> LDS per lane, tracked by ASYNCcnt.
__device__ __forceinline__ void async_cp16(const __bf16* g, __bf16* l) {
    unsigned loff = (unsigned)(size_t)l;
    unsigned long long ga = (unsigned long long)(size_t)g;
    asm volatile("global_load_async_to_lds_b128 %0, %1, off"
                 :: "v"(loff), "v"(ga) : "memory");
}
#define WAIT_ASYNC(n) asm volatile("s_wait_asynccnt " #n ::: "memory")

// ---------------------------------------------------------------------------
// Pre-pass 1: elementwise f32 -> bf16 (8 elems / thread)
// ---------------------------------------------------------------------------
__global__ void __launch_bounds__(256)
convert_bf16(const float* __restrict__ src, __bf16* __restrict__ dst) {
    const size_t i = ((size_t)blockIdx.x * 256 + threadIdx.x) * 8;
    const float4 a = *(const float4*)(src + i);
    const float4 b = *(const float4*)(src + i + 4);
    v8bf o;
    o[0] = to_bf16(a.x); o[1] = to_bf16(a.y); o[2] = to_bf16(a.z); o[3] = to_bf16(a.w);
    o[4] = to_bf16(b.x); o[5] = to_bf16(b.y); o[6] = to_bf16(b.z); o[7] = to_bf16(b.w);
    *(v8bf*)(dst + i) = o;
}

// ---------------------------------------------------------------------------
// Pre-pass 2: W[1024][1024] f32 -> Wt[1024][1024] bf16 transposed ([n][k])
// ---------------------------------------------------------------------------
#define DW 1024
__global__ void __launch_bounds__(256)
transpose_convert(const float* __restrict__ W, __bf16* __restrict__ Wt) {
    __shared__ __bf16 tile[64 * 72];
    const int tid = threadIdx.x;
    const int n0 = blockIdx.x * 64, k0 = blockIdx.y * 64;
#pragma unroll
    for (int i = 0; i < 4; ++i) {
        int idx = tid + i * 256;
        int r = idx >> 4, c = (idx & 15) << 2;
        const float4 w4 = *(const float4*)(W + (size_t)(k0 + r) * DW + n0 + c);
        tile[(c + 0) * 72 + r] = to_bf16(w4.x);
        tile[(c + 1) * 72 + r] = to_bf16(w4.y);
        tile[(c + 2) * 72 + r] = to_bf16(w4.z);
        tile[(c + 3) * 72 + r] = to_bf16(w4.w);
    }
    __syncthreads();
#pragma unroll
    for (int i = 0; i < 2; ++i) {
        int idx = tid + i * 256;
        int r = idx >> 3, c8 = (idx & 7) << 3;
        *(v8bf*)(Wt + (size_t)(n0 + r) * DW + k0 + c8) = *(const v8bf*)&tile[r * 72 + c8];
    }
}

// ---------------------------------------------------------------------------
// GEMM: C[4096x1024] = A_bf * Wt_bf + bias, optional output scale (Q: 0.125).
// Block tile 128x256, 8 waves in 2x4, each wave 64x64 (16 WMMA / K-step).
// Double-buffered async LDS staging (6 copies/thread/tile).
// ---------------------------------------------------------------------------
#define GM 4096
#define GN 1024
#define GK 1024
#define BM 128
#define BN 256
#define BK 32
#define LDT (BK + 8)   // 40 elems = 80B stride, 16B-aligned chunks

template <bool OUT_BF16>
__global__ void __launch_bounds__(256)
gemm_bf16(const __bf16* __restrict__ A, const __bf16* __restrict__ Wt,
          const float* __restrict__ bias, float oscale,
          float* __restrict__ Cf, __bf16* __restrict__ Cb) {
    __shared__ __bf16 As[2][BM * LDT];   // 2 x 10 KB
    __shared__ __bf16 Bs[2][BN * LDT];   // 2 x 20 KB

    const int tid  = threadIdx.x;
    const int lane = tid & 31;
    const int wave = tid >> 5;
    const int m0 = blockIdx.y * BM;
    const int n0 = blockIdx.x * BN;
    const int wm = (wave >> 2) * 64;     // 0 or 64
    const int wn = (wave & 3)  * 64;     // 0,64,128,192

    const int arow = lane & 15;
    const int koff = (lane >> 4) * 8;
    const int kbh  = (lane >> 4) * 16;
    const int hi8  = (lane >> 4) * 8;

    const int sr = tid >> 2;             // 0..63
    const int sc = (tid & 3) << 3;       // chunk col {0,8,16,24}

    auto stage = [&](int kt, int buf) {
        const int k0 = kt * BK;
#pragma unroll
        for (int i = 0; i < 2; ++i) {    // A: 128 rows
            const int r = sr + i * 64;
            async_cp16(A + (size_t)(m0 + r) * GK + k0 + sc, &As[buf][r * LDT + sc]);
        }
#pragma unroll
        for (int i = 0; i < 4; ++i) {    // Wt: 256 rows
            const int r = sr + i * 64;
            async_cp16(Wt + (size_t)(n0 + r) * GK + k0 + sc, &Bs[buf][r * LDT + sc]);
        }
    };

    v8f acc[4][4];
#pragma unroll
    for (int mi = 0; mi < 4; ++mi)
#pragma unroll
        for (int ni = 0; ni < 4; ++ni)
#pragma unroll
            for (int r = 0; r < 8; ++r) acc[mi][ni][r] = 0.0f;

    stage(0, 0);
    for (int kt = 0; kt < GK / BK; ++kt) {
        const int buf = kt & 1;
        if (kt + 1 < GK / BK) {
            stage(kt + 1, buf ^ 1);
            WAIT_ASYNC(6);               // current buffer's 6 copies done
        } else {
            WAIT_ASYNC(0);
        }
        __syncthreads();

        v16bf af[4];
#pragma unroll
        for (int mi = 0; mi < 4; ++mi) {
            const __bf16* p = &As[buf][(wm + mi * 16 + arow) * LDT + koff];
            af[mi] = ld_frag(p, p + 16);
        }
#pragma unroll
        for (int ni = 0; ni < 4; ++ni) {
            const __bf16* p = &Bs[buf][(wn + ni * 16 + (lane & 15)) * LDT + koff];
            v16bf bfr = ld_frag(p, p + 16);
#pragma unroll
            for (int mi = 0; mi < 4; ++mi)
                acc[mi][ni] = wmma_bf16(af[mi], bfr, acc[mi][ni]);
        }
        __syncthreads();
    }

#pragma unroll
    for (int ni = 0; ni < 4; ++ni) {
        const int col = n0 + wn + ni * 16 + (lane & 15);
        const float bv = bias[col];
#pragma unroll
        for (int mi = 0; mi < 4; ++mi)
#pragma unroll
            for (int r = 0; r < 8; ++r) {
                const int row = m0 + wm + mi * 16 + r + hi8;
                const float v = (acc[mi][ni][r] + bv) * oscale;
                if (OUT_BF16) Cb[(size_t)row * GN + col] = to_bf16(v);
                else          Cf[(size_t)row * GN + col] = v;
            }
    }
}

// ---------------------------------------------------------------------------
// Flash attention over bf16 Q/K/V (Q pre-scaled by 1/sqrt(dk)). Causal.
// ---------------------------------------------------------------------------
#define BT 2048
#define BD 1024
#define QB 128
#define KBK 64
#define DKH 64
#define LDH (DKH + 8)

__global__ void __launch_bounds__(256)
attn_flash(const __bf16* __restrict__ Q, const __bf16* __restrict__ Kg,
           const __bf16* __restrict__ Vg, __bf16* __restrict__ ctx) {
    __shared__ __bf16 Qs[QB * LDH];
    __shared__ __bf16 Ks[KBK * LDH];        // [key][dk] : B-frag layout for QK^T
    __shared__ __bf16 Vt[DKH * LDH];        // [dk][key] : B-frag layout for P*V
    __shared__ __bf16 Ps[8 * 16 * LDH];     // wave-private P staging

    const int tid  = threadIdx.x;
    const int lane = tid & 31;
    const int wave = tid >> 5;
    const int qb0 = blockIdx.x * QB;
    const int b   = blockIdx.y >> 4;
    const int h   = blockIdx.y & 15;
    const size_t rowbase = (size_t)b * BT * BD + (size_t)h * DKH;

    const int arow = lane & 15;
    const int koff = (lane >> 4) * 8;
    const int kbh  = (lane >> 4) * 16;
    const int hi8  = (lane >> 4) * 8;
    const int qw0  = qb0 + wave * 16;       // wave's first query row

    // stage Q (128x64 bf16) via async copies
#pragma unroll
    for (int i = 0; i < 4; ++i) {
        int idx = tid + i * 256;
        int r = idx >> 3, c8 = (idx & 7) << 3;
        async_cp16(Q + rowbase + (size_t)(qb0 + r) * BD + c8, &Qs[r * LDH + c8]);
    }
    WAIT_ASYNC(0);
    __syncthreads();

    v16bf qa[2];
#pragma unroll
    for (int c = 0; c < 2; ++c) {
        const __bf16* p = &Qs[(wave * 16 + arow) * LDH + c * 32 + koff];
        qa[c] = ld_frag(p, p + 16);
    }

    float mrow[8], lrow[8];
    v8f o[4];
#pragma unroll
    for (int r = 0; r < 8; ++r) { mrow[r] = -1e30f; lrow[r] = 0.0f; }
#pragma unroll
    for (int d = 0; d < 4; ++d)
#pragma unroll
        for (int r = 0; r < 8; ++r) o[d][r] = 0.0f;

    const int kb_end = qb0 / KBK + 2;
    for (int kb = 0; kb < kb_end; ++kb) {
        const int k0 = kb * KBK;
        __syncthreads();

        // K: async pure copy; V: load + transposed LDS store
#pragma unroll
        for (int i = 0; i < 2; ++i) {
            int idx = tid + i * 256;
            int r = idx >> 3, c8 = (idx & 7) << 3;
            async_cp16(Kg + rowbase + (size_t)(k0 + r) * BD + c8, &Ks[r * LDH + c8]);
            const v8bf v8 = *(const v8bf*)(Vg + rowbase + (size_t)(k0 + r) * BD + c8);
#pragma unroll
            for (int j = 0; j < 8; ++j) Vt[(c8 + j) * LDH + r] = v8[j];
        }
        if (kb + 1 < kb_end) {
            const size_t nr = rowbase + (size_t)(k0 + KBK + (tid >> 1)) * BD + ((tid & 1) << 5);
            __builtin_prefetch(Kg + nr, 0, 1);
            __builtin_prefetch(Vg + nr, 0, 1);
        }
        WAIT_ASYNC(0);
        __syncthreads();

        // S = (Q/sqrt(dk)) * K^T  -- scale already folded into Q
        v8f s[4];
#pragma unroll
        for (int nt = 0; nt < 4; ++nt) {
#pragma unroll
            for (int r = 0; r < 8; ++r) s[nt][r] = 0.0f;
            const __bf16* kp = &Ks[(nt * 16 + (lane & 15)) * LDH];
#pragma unroll
            for (int c = 0; c < 2; ++c) {
                v16bf bfr = ld_frag(kp + c * 32 + koff, kp + c * 32 + koff + 16);
                s[nt] = wmma_bf16(qa[c], bfr, s[nt]);
            }
        }

        // causal mask only on diagonal-intersecting blocks (wave-uniform)
        if (k0 + KBK - 1 > qw0) {
#pragma unroll
            for (int nt = 0; nt < 4; ++nt) {
                const int col = k0 + nt * 16 + (lane & 15);
#pragma unroll
                for (int r = 0; r < 8; ++r) {
                    const int row = qw0 + r + hi8;
                    if (col > row) s[nt][r] = -1e30f;
                }
            }
        }

        // online softmax (row spans 16 lanes of a half-wave)
#pragma unroll
        for (int r = 0; r < 8; ++r) {
            float mx = fmaxf(fmaxf(s[0][r], s[1][r]), fmaxf(s[2][r], s[3][r]));
#pragma unroll
            for (int off = 8; off >= 1; off >>= 1)
                mx = fmaxf(mx, __shfl_xor(mx, off, 32));
            const float mnew  = fmaxf(mrow[r], mx);
            const float alpha = __expf(mrow[r] - mnew);
            mrow[r] = mnew;
            float rs = 0.0f;
#pragma unroll
            for (int nt = 0; nt < 4; ++nt) {
                const float e = __expf(s[nt][r] - mnew);
                s[nt][r] = e;
                rs += e;
            }
#pragma unroll
            for (int off = 8; off >= 1; off >>= 1)
                rs += __shfl_xor(rs, off, 32);
            lrow[r] = lrow[r] * alpha + rs;
#pragma unroll
            for (int d = 0; d < 4; ++d) o[d][r] *= alpha;
        }

        // P -> wave-private LDS (D layout -> A layout)
        __bf16* pw = &Ps[wave * 16 * LDH];
#pragma unroll
        for (int nt = 0; nt < 4; ++nt)
#pragma unroll
            for (int r = 0; r < 8; ++r)
                pw[(r + hi8) * LDH + nt * 16 + (lane & 15)] = to_bf16(s[nt][r]);
        asm volatile("s_wait_dscnt 0" ::: "memory");

        v16bf pa[2];
#pragma unroll
        for (int c = 0; c < 2; ++c) {
            const __bf16* pp = &pw[arow * LDH + c * 32 + koff];
            pa[c] = ld_frag(pp, pp + 16);
        }

        // O += P * V
#pragma unroll
        for (int dt = 0; dt < 4; ++dt) {
            const __bf16* vp = &Vt[(dt * 16 + (lane & 15)) * LDH];
#pragma unroll
            for (int c = 0; c < 2; ++c) {
                v16bf bfr = ld_frag(vp + c * 32 + koff, vp + c * 32 + koff + 16);
                o[dt] = wmma_bf16(pa[c], bfr, o[dt]);
            }
        }
    }

    // normalize + store bf16 context
#pragma unroll
    for (int r = 0; r < 8; ++r) lrow[r] = 1.0f / lrow[r];
#pragma unroll
    for (int dt = 0; dt < 4; ++dt)
#pragma unroll
        for (int r = 0; r < 8; ++r) {
            const int row = qw0 + r + hi8;
            ctx[rowbase + (size_t)row * BD + dt * 16 + (lane & 15)] =
                to_bf16(o[dt][r] * lrow[r]);
        }
}

// ---------------------------------------------------------------------------
// launch
// ws (bf16 elems): xb 4M | 4x Wt 1M | Q 4M | K 4M | V 4M | ctx 4M = 48 MB
// ---------------------------------------------------------------------------
extern "C" void kernel_launch(void* const* d_in, const int* in_sizes, int n_in,
                              void* d_out, int out_size, void* d_ws, size_t ws_size,
                              hipStream_t stream) {
    (void)in_sizes; (void)n_in; (void)out_size; (void)ws_size;
    const float* x  = (const float*)d_in[0];
    const float* Wq = (const float*)d_in[1];
    const float* bq = (const float*)d_in[2];
    const float* Wk = (const float*)d_in[3];
    const float* bk = (const float*)d_in[4];
    const float* Wv = (const float*)d_in[5];
    const float* bv = (const float*)d_in[6];
    const float* Wo = (const float*)d_in[7];
    const float* bo = (const float*)d_in[8];
    float* out = (float*)d_out;

    __bf16* ws = (__bf16*)d_ws;
    const size_t NT = (size_t)GM * GN;
    const size_t WW = (size_t)GK * GN;
    __bf16* xb  = ws;
    __bf16* Wqt = xb  + NT;
    __bf16* Wkt = Wqt + WW;
    __bf16* Wvt = Wkt + WW;
    __bf16* Wot = Wvt + WW;
    __bf16* Qb  = Wot + WW;
    __bf16* Kb  = Qb  + NT;
    __bf16* Vb  = Kb  + NT;
    __bf16* Cb  = Vb  + NT;

    dim3 blk(256);
    convert_bf16<<<dim3(NT / (256 * 8)), blk, 0, stream>>>(x, xb);
    dim3 tgrid(DW / 64, DW / 64);
    transpose_convert<<<tgrid, blk, 0, stream>>>(Wq, Wqt);
    transpose_convert<<<tgrid, blk, 0, stream>>>(Wk, Wkt);
    transpose_convert<<<tgrid, blk, 0, stream>>>(Wv, Wvt);
    transpose_convert<<<tgrid, blk, 0, stream>>>(Wo, Wot);

    dim3 gemm_grid(GN / BN, GM / BM);        // (4, 32)
    gemm_bf16<true ><<<gemm_grid, blk, 0, stream>>>(xb, Wqt, bq, 0.125f, nullptr, Qb);
    gemm_bf16<true ><<<gemm_grid, blk, 0, stream>>>(xb, Wkt, bk, 1.0f,   nullptr, Kb);
    gemm_bf16<true ><<<gemm_grid, blk, 0, stream>>>(xb, Wvt, bv, 1.0f,   nullptr, Vb);

    dim3 attn_grid(BT / QB, 2 * 16);         // (16, B*H=32)
    attn_flash<<<attn_grid, blk, 0, stream>>>(Qb, Kb, Vb, Cb);

    gemm_bf16<false><<<gemm_grid, blk, 0, stream>>>(Cb, Wot, bo, 1.0f, out, nullptr);
}
